// AttentionDMNtied_59820304499276
// MI455X (gfx1250) — matrix operand order repaired and checked by reading, not compiled, not run on hardware
//
#include <hip/hip_runtime.h>
#include <math.h>

// ---------------- WMMA types (gfx1250, wave32) ----------------
typedef __attribute__((ext_vector_type(16))) __bf16 v16bf;
typedef __attribute__((ext_vector_type(8)))  __bf16 v8bf;
typedef __attribute__((ext_vector_type(2)))  __bf16 v2bf;
typedef __attribute__((ext_vector_type(8)))  float  v8f;

#define BM 128
#define BN 64
#define BK 32
#define LDA 40   // bf16 units per A row (80B, multiple of 16B)
#define LDB 40   // bf16 units per Bt row

struct GemmP {
  const float* A;  long long as_m, as_k, as_b;
  const float* Bm; long long bs_k, bs_n, bs_b;
  float* C;        long long cs_m, cs_n, cs_b;
  const float* bias;            // per-output-column (n), nullable
  int M, N, K;
  int act;                      // 0 none, 1 relu, 2 tanh
  int amode;                    // 0 plain A, 1 fused z = [f*q, f*m, |f-q|, |f-m|]
  const float* zq; const float* zpm; int zB; // zB must be pow2
};

// Strided GEMM: C[m,n] = act( sum_k A[m,k]*B[k,n] + bias[n] )
// 128x64 block tile, BK=32, double-buffered LDS, bf16 staging.
// 8 wave32s; each wave computes a 32x32 subtile with 4x v_wmma_f32_16x16x32_bf16.
// Fragments are loaded as 2x ds_load_b128 per matrix (ISA 16-bit A/B lane layouts).
__global__ __launch_bounds__(256) void gemm_wmma_bf16(GemmP p) {
  __shared__ __align__(16) __bf16 As[2][BM * LDA];
  __shared__ __align__(16) __bf16 Bs[2][BN * LDB];   // stored transposed: [n][k]

  const int tid   = threadIdx.x;
  const int lane  = tid & 31;            // wave32
  const int wave  = tid >> 5;
  const int sr    = (wave >> 1) * 32;    // wave row base (4 row-waves)
  const int scb   = (wave & 1) * 32;     // wave col base (2 col-waves)
  const int mrow  = lane & 15;
  const int khalf = lane >> 4;

  const long long bz = blockIdx.z;
  const float* Ab = p.A  + bz * p.as_b;
  const float* Bb = p.Bm + bz * p.bs_b;
  float*       Cb = p.C  + bz * p.cs_b;
  const int m0 = blockIdx.y * BM;
  const int n0 = blockIdx.x * BN;
  const int Ksub = p.amode ? (p.K >> 2) : 1;

  v8f acc[4] = {{}, {}, {}, {}};

  // ---- stage A tile (BM x BK) as bf16 pairs; fused-z selection is uniform/tile
  auto stageA = [&](int buf, int kt) {
    const float* zp = p.zq;
    bool prod = true;
    int mmb = kt;
    if (p.amode) {
      int j = kt / Ksub;                 // uniform scalar per tile
      mmb = kt - j * Ksub;
      zp = (j & 1) ? p.zpm : p.zq;
      prod = (j < 2);
    }
    #pragma unroll
    for (int i = 0; i < (BM * BK / 2) / 256; ++i) {   // 8 iters
      int pe = i * 256 + tid;
      int r  = pe >> 4;
      int kp = (pe & 15) * 2;
      int gm = m0 + r;
      float a0 = 0.0f, a1 = 0.0f;
      if (gm < p.M) {
        if (p.amode == 0) {
          int gk = kt + kp;
          const float* ar = Ab + (long long)gm * p.as_m;
          if (gk < p.K)     a0 = ar[(long long)gk * p.as_k];
          if (gk + 1 < p.K) a1 = ar[(long long)(gk + 1) * p.as_k];
        } else {
          int mm = mmb + kp;
          int bb = gm & (p.zB - 1);
          const float* ar = Ab + (long long)gm * p.as_m;
          const float* zr = zp + (long long)bb * Ksub;
          float f0 = ar[(long long)mm * p.as_k];
          float f1 = ar[(long long)(mm + 1) * p.as_k];
          float o0 = zr[mm], o1 = zr[mm + 1];
          a0 = prod ? (f0 * o0) : fabsf(f0 - o0);
          a1 = prod ? (f1 * o1) : fabsf(f1 - o1);
        }
      }
      v2bf pk; pk[0] = (__bf16)a0; pk[1] = (__bf16)a1;
      *(v2bf*)(&As[buf][r * LDA + kp]) = pk;
    }
  };

  // ---- stage B tile transposed (BN x BK) ----
  auto stageB = [&](int buf, int kt) {
    #pragma unroll
    for (int i = 0; i < (BN * BK / 2) / 256; ++i) {   // 4 iters
      int pe = i * 256 + tid;
      int n  = pe >> 4;
      int kp = (pe & 15) * 2;
      int gn = n0 + n;
      int gk = kt + kp;
      float b0 = 0.0f, b1 = 0.0f;
      if (gn < p.N) {
        long long coln = (long long)gn * p.bs_n;
        if (gk < p.K)     b0 = Bb[(long long)gk * p.bs_k + coln];
        if (gk + 1 < p.K) b1 = Bb[(long long)(gk + 1) * p.bs_k + coln];
      }
      v2bf pk; pk[0] = (__bf16)b0; pk[1] = (__bf16)b1;
      *(v2bf*)(&Bs[buf][n * LDB + kp]) = pk;
    }
  };

  // ---- one K-slab of compute: 4 fragments, 4 WMMAs ----
  auto compute = [&](int buf) {
    const __bf16* Ash = As[buf];
    const __bf16* Bsh = Bs[buf];
    v16bf af[2], bfv[2];
    #pragma unroll
    for (int a = 0; a < 2; ++a) {
      int row = sr + a * 16 + mrow;
      v8bf lo = *(const v8bf*)(Ash + row * LDA + khalf * 8);
      v8bf hi = *(const v8bf*)(Ash + row * LDA + 16 + khalf * 8);
      af[a] = __builtin_shufflevector(lo, hi, 0,1,2,3,4,5,6,7,8,9,10,11,12,13,14,15);
    }
    #pragma unroll
    for (int b = 0; b < 2; ++b) {
      int nr = scb + b * 16 + mrow;
      v8bf lo = *(const v8bf*)(Bsh + nr * LDB + khalf * 8);
      v8bf hi = *(const v8bf*)(Bsh + nr * LDB + 16 + khalf * 8);
      bfv[b] = __builtin_shufflevector(lo, hi, 0,1,2,3,4,5,6,7,8,9,10,11,12,13,14,15);
    }
    #pragma unroll
    for (int a = 0; a < 2; ++a)
      #pragma unroll
      for (int b = 0; b < 2; ++b)
        acc[a * 2 + b] = __builtin_amdgcn_wmma_f32_16x16x32_bf16(
            false, af[a], false, bfv[b], (short)0, acc[a * 2 + b], false, false);
  };

  // ---- double-buffered main loop ----
  const int nT = (p.K + BK - 1) / BK;
  stageA(0, 0); stageB(0, 0);
  __syncthreads();
  for (int t = 0; t < nT; ++t) {
    int cur = t & 1;
    if (t + 1 < nT) { stageA(cur ^ 1, (t + 1) * BK); stageB(cur ^ 1, (t + 1) * BK); }
    compute(cur);
    __syncthreads();
  }

  // ---- epilogue: C/D layout VGPR r -> M=r (lanes 0-15), M=r+8 (lanes 16-31)
  const int mhalf = khalf * 8;
  #pragma unroll
  for (int a = 0; a < 2; ++a) {
    #pragma unroll
    for (int b = 0; b < 2; ++b) {
      int gn = n0 + scb + b * 16 + mrow;
      float bv = (p.bias && gn < p.N) ? p.bias[gn] : 0.0f;
      #pragma unroll
      for (int r = 0; r < 8; ++r) {
        int gm = m0 + sr + a * 16 + mhalf + r;
        if (gm < p.M && gn < p.N) {
          float v = acc[a * 2 + b][r] + bv;
          if (p.act == 1) v = fmaxf(v, 0.0f);
          else if (p.act == 2) v = tanhf(v);
          Cb[(long long)gm * p.cs_m + (long long)gn * p.cs_n] = v;
        }
      }
    }
  }
}

// ---------------- elementwise helpers ----------------
__device__ __forceinline__ float sigm(float x) { return 1.0f / (1.0f + expf(-x)); }

__global__ void k_zero(float* p, long long n) {
  long long i = (long long)blockIdx.x * blockDim.x + threadIdx.x;
  if (i < n) p[i] = 0.0f;
}
__global__ void k_copy(float* d, const float* s, long long n) {
  long long i = (long long)blockIdx.x * blockDim.x + threadIdx.x;
  if (i < n) d[i] = s[i];
}

// GRU step: h = (1-z)*n + z*h ; writes/accumulates into facts row
__global__ void k_gru_step(const float* __restrict__ xp, const float* __restrict__ hp,
                           float* __restrict__ h, float* __restrict__ facts,
                           int addmode, int Bn, int Mm) {
  int i = blockIdx.x * blockDim.x + threadIdx.x;
  if (i >= Bn * Mm) return;
  int b = i / Mm, m = i - b * Mm;
  const float* xr = xp + (long long)b * 3 * Mm;
  const float* hr = hp + (long long)b * 3 * Mm;
  float r = sigm(xr[m] + hr[m]);
  float z = sigm(xr[Mm + m] + hr[Mm + m]);
  float n = tanhf(xr[2 * Mm + m] + r * hr[2 * Mm + m]);
  float hn = (1.0f - z) * n + z * h[i];
  h[i] = hn;
  if (addmode) facts[i] += hn; else facts[i] = hn;
}

// AGRU step: C = g*tanh(fw + r*(C@U^T+Ub)) + (1-g)*C, r = sigm(fr + C@Ur^T+Urb)
__global__ void k_agru_step(const float* __restrict__ fr, const float* __restrict__ fw,
                            const float* __restrict__ cr1, const float* __restrict__ cr2,
                            float* __restrict__ C, const float* __restrict__ G,
                            int Bn, int Mm) {
  int i = blockIdx.x * blockDim.x + threadIdx.x;
  if (i >= Bn * Mm) return;
  int b = i / Mm;
  float r  = sigm(fr[i] + cr1[i]);
  float ht = tanhf(fw[i] + r * cr2[i]);
  float g  = G[b];
  C[i] = g * ht + (1.0f - g) * C[i];
}

// attention scores: s[row] = g1[row,:] . z2w + z2b
__global__ void k_scores(const float* __restrict__ g1, const float* __restrict__ z2w,
                         const float* __restrict__ z2b, float* __restrict__ s,
                         long long rows, int Mm) {
  long long row = (long long)blockIdx.x * blockDim.x + threadIdx.x;
  if (row >= rows) return;
  const float* gr = g1 + row * Mm;
  float acc = 0.0f;
  for (int h = 0; h < Mm; ++h) acc += gr[h] * z2w[h];
  s[row] = acc + z2b[0];
}

// softmax over t for each b; s and G laid out as [t*B + b]
__global__ void k_softmax(const float* __restrict__ s, float* __restrict__ G, int T, int Bn) {
  __shared__ float red[256];
  int b = blockIdx.x, t = threadIdx.x;
  float v = (t < T) ? s[(long long)t * Bn + b] : -1e30f;
  red[t] = v; __syncthreads();
  for (int o = 128; o > 0; o >>= 1) { if (t < o) red[t] = fmaxf(red[t], red[t + o]); __syncthreads(); }
  float mx = red[0]; __syncthreads();
  float e = (t < T) ? expf(v - mx) : 0.0f;
  red[t] = e; __syncthreads();
  for (int o = 128; o > 0; o >>= 1) { if (t < o) red[t] += red[t + o]; __syncthreads(); }
  float sum = red[0];
  if (t < T) G[(long long)t * Bn + b] = e / sum;
}

__global__ void k_concat(float* __restrict__ dst, const float* __restrict__ mprev,
                         const float* __restrict__ C, const float* __restrict__ q,
                         int Bn, int Mm) {
  int i = blockIdx.x * blockDim.x + threadIdx.x;
  if (i >= Bn * 3 * Mm) return;
  int b = i / (3 * Mm), j = i - b * 3 * Mm;
  float v;
  if (j < Mm)            v = mprev[b * Mm + j];
  else if (j < 2 * Mm)   v = C[b * Mm + (j - Mm)];
  else                   v = q[b * Mm + (j - 2 * Mm)];
  dst[i] = v;
}

// out[b,m] = m[b,m] * sum_t facts[t,b,m]
__global__ void k_final(float* __restrict__ out, const float* __restrict__ mb,
                        const float* __restrict__ facts, int T, int Bn, int Mm) {
  int i = blockIdx.x * blockDim.x + threadIdx.x;
  if (i >= Bn * Mm) return;
  float s = 0.0f;
  for (int t = 0; t < T; ++t) s += facts[(long long)t * Bn * Mm + i];
  out[i] = mb[i] * s;
}

// ---------------- driver ----------------
extern "C" void kernel_launch(void* const* d_in, const int* in_sizes, int n_in,
                              void* d_out, int out_size, void* d_ws, size_t ws_size,
                              hipStream_t stream) {
  (void)in_sizes; (void)n_in; (void)out_size;
  const int Bsz = 256, Vc = 2048, HWt = 196, Qd = 1024, Md = 512, Tt = 196, Ll = 3;

  const float* vis   = (const float*)d_in[0];
  const float* ques  = (const float*)d_in[1];
  const float* convw = (const float*)d_in[2];
  const float* linw  = (const float*)d_in[3];
  const float* linb  = (const float*)d_in[4];
  const float* wi_f  = (const float*)d_in[5];
  const float* wh_f  = (const float*)d_in[6];
  const float* bi_f  = (const float*)d_in[7];
  const float* bh_f  = (const float*)d_in[8];
  const float* wi_b  = (const float*)d_in[9];
  const float* wh_b  = (const float*)d_in[10];
  const float* bi_b  = (const float*)d_in[11];
  const float* bh_b  = (const float*)d_in[12];
  const float* Wr_w  = (const float*)d_in[13];
  const float* Wr_b  = (const float*)d_in[14];
  const float* Ur_w  = (const float*)d_in[15];
  const float* Ur_b  = (const float*)d_in[16];
  const float* Wn_w  = (const float*)d_in[17];
  const float* Wn_b  = (const float*)d_in[18];
  const float* U_w   = (const float*)d_in[19];
  const float* U_b   = (const float*)d_in[20];
  const float* z1w   = (const float*)d_in[21];
  const float* z1b   = (const float*)d_in[22];
  const float* z2w   = (const float*)d_in[23];
  const float* z2b   = (const float*)d_in[24];
  const float* nmw   = (const float*)d_in[25];
  const float* nmb   = (const float*)d_in[26];
  float* out = (float*)d_out;

  const long long NTB = (long long)Tt * Bsz;       // 50176 (t,b) rows
  const long long SZ1 = NTB * Md;                  // 25,690,112 floats
  const long long needF = 7 * SZ1 + 2LL * 1024 * 1024;
  if (ws_size < (size_t)(needF * 4)) return;       // scratch too small

  float* ws    = (float*)d_ws;
  float* facts = ws;                               // x, later facts (T*B, M)
  float* xpf   = ws + SZ1;                         // (T*B, 3M)
  float* xpb   = xpf + 3 * SZ1;                    // (T*B, 3M)
  float* smp   = xpb + 3 * SZ1;
  float* qb  = smp; smp += (long long)Bsz * Md;
  float* mb  = smp; smp += (long long)Bsz * Md;
  float* hb  = smp; smp += (long long)Bsz * Md;
  float* hp  = smp; smp += (long long)Bsz * 3 * Md;
  float* cr1 = smp; smp += (long long)Bsz * Md;
  float* cr2 = smp; smp += (long long)Bsz * Md;
  float* Cb  = smp; smp += (long long)Bsz * Md;
  float* cat = smp; smp += (long long)Bsz * 3 * Md;
  float* sc  = smp; smp += NTB;
  float* Gb  = smp; smp += NTB;
  float* g1  = xpf;                 // phase-6 reuse of xp regions
  float* fr  = xpf + SZ1;
  float* fw  = xpf + 2 * SZ1;

  auto gemm = [&](const float* A, long long am, long long ak, long long ab,
                  const float* Bp, long long bk, long long bn, long long bb,
                  float* Cp, long long cm, long long cn, long long cb,
                  const float* bias, int Mq, int Nq, int Kq, int act, int batch,
                  int amode, const float* zq, const float* zpm, int zB) {
    GemmP p{A, am, ak, ab, Bp, bk, bn, bb, Cp, cm, cn, cb,
            bias, Mq, Nq, Kq, act, amode, zq, zpm, zB};
    dim3 g((Nq + BN - 1) / BN, (Mq + BM - 1) / BM, batch);
    gemm_wmma_bf16<<<g, dim3(256), 0, stream>>>(p);
  };
  const int EW = (Bsz * Md + 255) / 256;           // 512 blocks for (B,M) elementwise

  // 1) conv 1x1 + relu -> x[(t*B+b), m]   (C[m, t] per batch b)
  gemm(convw, Vc, 1, 0,
       vis, HWt, 1, (long long)Vc * HWt,
       facts, 1, (long long)Bsz * Md, Md,
       nullptr, Md, HWt, Vc, /*relu*/1, Bsz, 0, nullptr, nullptr, 0);

  // 2) q = relu(question @ lin_w^T + lin_b)
  gemm(ques, Qd, 1, 0, linw, 1, Qd, 0, qb, Md, 1, 0,
       linb, Bsz, Md, Qd, 1, 1, 0, nullptr, nullptr, 0);

  // 3) GRU input projections for both directions (batched over all t)
  gemm(facts, Md, 1, 0, wi_f, 1, Md, 0, xpf, 3 * Md, 1, 0,
       bi_f, (int)NTB, 3 * Md, Md, 0, 1, 0, nullptr, nullptr, 0);
  gemm(facts, Md, 1, 0, wi_b, 1, Md, 0, xpb, 3 * Md, 1, 0,
       bi_b, (int)NTB, 3 * Md, Md, 0, 1, 0, nullptr, nullptr, 0);

  // 4) forward GRU scan (writes facts rows)
  k_zero<<<EW, 256, 0, stream>>>(hb, (long long)Bsz * Md);
  for (int t = 0; t < Tt; ++t) {
    gemm(hb, Md, 1, 0, wh_f, 1, Md, 0, hp, 3 * Md, 1, 0,
         bh_f, Bsz, 3 * Md, Md, 0, 1, 0, nullptr, nullptr, 0);
    k_gru_step<<<EW, 256, 0, stream>>>(xpf + (long long)t * Bsz * 3 * Md, hp, hb,
                                       facts + (long long)t * Bsz * Md, 0, Bsz, Md);
  }
  // 5) backward GRU scan (accumulates into facts rows)
  k_zero<<<EW, 256, 0, stream>>>(hb, (long long)Bsz * Md);
  for (int s = 0; s < Tt; ++s) {
    int t = Tt - 1 - s;
    gemm(hb, Md, 1, 0, wh_b, 1, Md, 0, hp, 3 * Md, 1, 0,
         bh_b, Bsz, 3 * Md, Md, 0, 1, 0, nullptr, nullptr, 0);
    k_gru_step<<<EW, 256, 0, stream>>>(xpb + (long long)t * Bsz * 3 * Md, hp, hb,
                                       facts + (long long)t * Bsz * Md, 1, Bsz, Md);
  }

  // 6) episodic memory, m0 = q
  k_copy<<<EW, 256, 0, stream>>>(mb, qb, (long long)Bsz * Md);
  for (int l = 0; l < Ll; ++l) {
    const float* Wr  = Wr_w + (long long)l * Md * Md;
    const float* Wrb = Wr_b + (long long)l * Md;
    const float* Ur  = Ur_w + (long long)l * Md * Md;
    const float* Urb = Ur_b + (long long)l * Md;
    const float* Wn  = Wn_w + (long long)l * Md * Md;
    const float* Wnb = Wn_b + (long long)l * Md;
    const float* Uu  = U_w  + (long long)l * Md * Md;
    const float* Uub = U_b  + (long long)l * Md;

    // g1 = tanh(z @ z1w^T + z1b), z built on the fly from facts/q/m (amode=1)
    gemm(facts, Md, 1, 0, z1w + (long long)l * Md * 4 * Md, 1, 4 * Md, 0,
         g1, Md, 1, 0, z1b + (long long)l * Md,
         (int)NTB, Md, 4 * Md, /*tanh*/2, 1, /*amode*/1, qb, mb, Bsz);
    k_scores<<<(int)((NTB + 255) / 256), 256, 0, stream>>>(
        g1, z2w + (long long)l * Md, z2b + l, sc, NTB, Md);
    k_softmax<<<Bsz, 256, 0, stream>>>(sc, Gb, Tt, Bsz);

    // fact projections (biases folded in)
    gemm(facts, Md, 1, 0, Wr, 1, Md, 0, fr, Md, 1, 0,
         Wrb, (int)NTB, Md, Md, 0, 1, 0, nullptr, nullptr, 0);
    gemm(facts, Md, 1, 0, Wn, 1, Md, 0, fw, Md, 1, 0,
         Wnb, (int)NTB, Md, Md, 0, 1, 0, nullptr, nullptr, 0);

    // AGRU scan
    k_zero<<<EW, 256, 0, stream>>>(Cb, (long long)Bsz * Md);
    for (int t = 0; t < Tt; ++t) {
      gemm(Cb, Md, 1, 0, Ur, 1, Md, 0, cr1, Md, 1, 0,
           Urb, Bsz, Md, Md, 0, 1, 0, nullptr, nullptr, 0);
      gemm(Cb, Md, 1, 0, Uu, 1, Md, 0, cr2, Md, 1, 0,
           Uub, Bsz, Md, Md, 0, 1, 0, nullptr, nullptr, 0);
      k_agru_step<<<EW, 256, 0, stream>>>(fr + (long long)t * Bsz * Md,
                                          fw + (long long)t * Bsz * Md,
                                          cr1, cr2, Cb, Gb + (long long)t * Bsz, Bsz, Md);
    }
    // m = relu(concat([m, C, q]) @ nmw^T + nmb)
    k_concat<<<(Bsz * 3 * Md + 255) / 256, 256, 0, stream>>>(cat, mb, Cb, qb, Bsz, Md);
    gemm(cat, 3 * Md, 1, 0, nmw + (long long)l * Md * 3 * Md, 1, 3 * Md, 0,
         mb, Md, 1, 0, nmb + (long long)l * Md,
         Bsz, Md, 3 * Md, /*relu*/1, 1, 0, nullptr, nullptr, 0);
  }

  // 7) out[b,m] = m[b,m] * sum_t facts[t,b,m]
  k_final<<<EW, 256, 0, stream>>>(out, mb, facts, Tt, Bsz, Md);
}
